// GAT_prompt_28424093565057
// MI455X (gfx1250) — matrix operand order repaired and checked by reading, not compiled, not run on hardware
//
#include <hip/hip_runtime.h>
#include <hip/hip_bf16.h>

// ---------------------------------------------------------------------------
// GAT forward for MI455X (gfx1250, wave32, WMMA).
//   * adj -> 2MB bitmask (L2-resident) via __ballot
//   * all B-operands pre-transposed once, hot loops are b128-only
//   * double-buffered LDS tiles staged with GLOBAL_LOAD_ASYNC_TO_LDS_B128
//     (ASYNCcnt) when the builtin exists; uint4 copy fallback otherwise
//   * flash-style masked softmax; P built directly in A-fragment registers
// ---------------------------------------------------------------------------

typedef __attribute__((ext_vector_type(16))) __bf16 bf16x16;
typedef __attribute__((ext_vector_type(8)))  float  floatx8;

union FragB { bf16x16 v; unsigned u[8]; uint4 q[2]; __bf16 h[16]; };
union FragC { floatx8 v; float f[8]; };

#define WMMA_BF16(a, b, c) \
  __builtin_amdgcn_wmma_f32_16x16x32_bf16(false, (a), false, (b), (short)0, (c), false, false)

// ---- async global->LDS staging (CDNA5 GLOBAL_LOAD_ASYNC_TO_LDS_B128) ------
#if defined(__has_builtin)
#if __has_builtin(__builtin_amdgcn_global_load_async_to_lds_b128)
#define HAVE_ASYNC_LDS 1
#endif
#endif
#ifndef HAVE_ASYNC_LDS
#define HAVE_ASYNC_LDS 0
#endif

typedef int v4i_gen __attribute__((vector_size(16)));  // matches builtin's V4i

__device__ inline void copy16_g2l(const uint4* gsrc, uint4* ldst) {
#if HAVE_ASYNC_LDS
  __builtin_amdgcn_global_load_async_to_lds_b128(
      (__attribute__((address_space(1))) v4i_gen*)gsrc,
      (__attribute__((address_space(3))) v4i_gen*)ldst, 0, 0);
#else
  *ldst = *gsrc;
#endif
}

__device__ inline void async_wait_all() {
#if HAVE_ASYNC_LDS
#if __has_builtin(__builtin_amdgcn_s_wait_asynccnt)
  __builtin_amdgcn_s_wait_asynccnt(0);
#else
  asm volatile("s_wait_asynccnt 0x0" ::: "memory");
#endif
#endif
}

// A-fragment (16x32 bf16), ISA 7.12.2 interleaved layout:
// lane L (r=L&15, hi=L>>4): u32 q holds K pair, K = hi*8 + (q&3)*2 + (q>=4?16:0)
// -> u32 indices {0..3, 8..11} relative to row base: two contiguous 16B runs.
__device__ inline bf16x16 lds_afrag(const __bf16* tile, int stride, int lane) {
  const int r = lane & 15, hi = lane >> 4;
  const __bf16* base = tile + r * stride + hi * 8;
  FragB f;
  f.q[0] = *(const uint4*)base;
  f.q[1] = *(const uint4*)(base + 16);
  return f.v;
}

// B-fragment (32x16 bf16), interleaved layout matching A-fragment:
// lane L: column n = L&15, element e -> K = (L>>4)*8 + (e&7) + (e>>3)*16
__device__ inline bf16x16 lds_bfrag(const __bf16* tile, int stride, int lane) {
  const int n = lane & 15, hi = lane >> 4;
  const __bf16* base = tile + n * stride + hi * 8;
  FragB f;
  f.q[0] = *(const uint4*)base;
  f.q[1] = *(const uint4*)(base + 16);
  return f.v;
}

// ---------------------------------------------------------------------------
// pack adjacency int32 -> bitmask, wave32 ballot
// ---------------------------------------------------------------------------
__global__ __launch_bounds__(256) void pack_mask_kernel(const int* __restrict__ adj,
                                                        unsigned* __restrict__ mb) {
  const int t = blockIdx.x * 256 + threadIdx.x;
  const int pred = adj[t] > 0;
  const unsigned long long b = __ballot(pred);
  if ((threadIdx.x & 31) == 0) mb[t >> 5] = (unsigned)b;
}

// ---------------------------------------------------------------------------
// f32 -> bf16 (row-major copy)
// ---------------------------------------------------------------------------
__global__ __launch_bounds__(256) void cvt_bf16_kernel(const float* __restrict__ in,
                                                       __bf16* __restrict__ out, int n) {
  const int i = blockIdx.x * 256 + threadIdx.x;
  if (i < n) out[i] = (__bf16)in[i];
}

// ---------------------------------------------------------------------------
// f32 [K][N] -> bf16 [N][K] (weight transpose+convert, one-shot, batched)
// ---------------------------------------------------------------------------
__global__ __launch_bounds__(256) void cvt_t_kernel(const float* __restrict__ in,
                                                    __bf16* __restrict__ out,
                                                    int K, int N,
                                                    long strideIn, long strideOut) {
  in  += (long)blockIdx.y * strideIn;
  out += (long)blockIdx.y * strideOut;
  const int idx = blockIdx.x * 256 + threadIdx.x;
  if (idx >= K * N) return;
  const int n = idx / K, k = idx % K;
  out[idx] = (__bf16)in[(long)k * N + n];
}

// ---------------------------------------------------------------------------
// bf16 [rows][D] -> bf16 [D][rows], LDS-tiled (64-row tiles), batched
// ---------------------------------------------------------------------------
template <int D>
__global__ __launch_bounds__(256) void transpose_bf16_kernel(
    const __bf16* __restrict__ in, __bf16* __restrict__ out, int rows,
    long strideIn, long strideOut) {
  __shared__ __bf16 t[64][D + 4];
  in  += (long)blockIdx.y * strideIn;
  out += (long)blockIdx.y * strideOut;
  const int tid = threadIdx.x;
  const int i0 = blockIdx.x * 64;
#pragma unroll
  for (int idx = tid; idx < 64 * D; idx += 256) {
    const int r = idx / D, c = idx % D;          // coalesced read
    t[r][c] = in[(long)(i0 + r) * D + c];
  }
  __syncthreads();
#pragma unroll
  for (int idx = tid; idx < 64 * D; idx += 256) {
    const int n = idx >> 6, ii = idx & 63;       // coalesced write
    out[(long)n * rows + i0 + ii] = t[ii][n];
  }
}

// ---------------------------------------------------------------------------
// Tiled bf16 WMMA GEMM. C[M, NF*16] = A[M,K] @ B[K,N], B given pre-transposed
// as WT[N][K]. Block: 256 threads = 8 waves, tile 128 x N, BK = 32,
// double-buffered async LDS staging. grid.y batches WT/Cb (attention heads).
// ---------------------------------------------------------------------------
template <int NF>
__global__ __launch_bounds__(256) void gemm_bf16_wmma(
    const __bf16* __restrict__ A, const __bf16* __restrict__ WT,
    int M, int K,
    float* __restrict__ Cf, __bf16* __restrict__ Cb,
    const float* __restrict__ bias,
    long strideB, long strideCb) {
  constexpr int N = NF * 16;
  constexpr int AS = 40;                       // As row stride (bf16), 80B
  constexpr int BS = 40;                       // BsT row stride (bf16), 80B
  __shared__ __align__(16) __bf16 As[2][128 * AS];
  __shared__ __align__(16) __bf16 BsT[2][64 * BS];

  const int tid  = threadIdx.x;
  const int lane = tid & 31;
  const int wave = tid >> 5;
  const int row0 = blockIdx.x * 128;
  WT += (long)blockIdx.y * strideB;
  if (Cb) Cb += (long)blockIdx.y * strideCb;

  FragC acc[NF];
#pragma unroll
  for (int f = 0; f < NF; ++f)
#pragma unroll
    for (int r = 0; r < 8; ++r) acc[f].f[r] = 0.f;

  const uint4* Aq  = (const uint4*)A;          // 8 bf16 per uint4
  const uint4* WTq = (const uint4*)WT;
  const int Kq = K >> 3;

  auto stage = [&](int kk, int b) {
    uint4* asq = (uint4*)As[b];
    uint4* bsq = (uint4*)BsT[b];
#pragma unroll
    for (int it = 0; it < 2; ++it) {
      const int idx = tid + it * 256;
      const int r = idx >> 2, c4 = idx & 3;
      copy16_g2l(Aq + (long)(row0 + r) * Kq + (kk >> 3) + c4,
                 asq + r * (AS / 8) + c4);
    }
#pragma unroll
    for (int idx = tid; idx < N * 4; idx += 256) {
      const int n = idx >> 2, c4 = idx & 3;
      copy16_g2l(WTq + (long)n * Kq + (kk >> 3) + c4,
                 bsq + n * (BS / 8) + c4);
    }
  };

  stage(0, 0);
  async_wait_all();
  __syncthreads();

  int buf = 0;
  for (int kk = 0; kk < K; kk += 32) {
    if (kk + 32 < K) stage(kk + 32, buf ^ 1);  // overlap next tile with WMMAs

    FragB af; af.v = lds_afrag(As[buf] + wave * 16 * AS, AS, lane);
#pragma unroll
    for (int nf = 0; nf < NF; ++nf) {
      FragB bf_; bf_.v = lds_bfrag(BsT[buf] + nf * 16 * BS, BS, lane);
      acc[nf].v = WMMA_BF16(af.v, bf_.v, acc[nf].v);
    }
    async_wait_all();
    __syncthreads();
    buf ^= 1;
  }

  // epilogue: C layout: lane -> col (lane&15), +8 rows for hi; VGPR r -> row
  const int col = lane & 15;
  const int rb  = (lane >> 4) * 8;
#pragma unroll
  for (int nf = 0; nf < NF; ++nf) {
    const int c = nf * 16 + col;
    const float bv = bias ? bias[c] : 0.f;
#pragma unroll
    for (int r = 0; r < 8; ++r) {
      const int row = row0 + wave * 16 + rb + r;
      if (row < M) {
        const float v = acc[nf].f[r] + bv;
        if (Cf) Cf[(long)row * N + c] = v;
        if (Cb) Cb[(long)row * N + c] = (__bf16)v;
      }
    }
  }
}

// ---------------------------------------------------------------------------
// f1 = Wh@a[:D], f2 = Wh@a[D:]  (Wh row-major bf16)
// ---------------------------------------------------------------------------
__global__ __launch_bounds__(256) void compute_f12_kernel(
    const __bf16* __restrict__ Wh, const float* __restrict__ a,
    int D, int rows, float* __restrict__ f1, float* __restrict__ f2,
    long whStride, int aStride, int fStride) {
  const int h = blockIdx.y;
  Wh += (long)h * whStride;
  a  += h * aStride;
  f1 += h * fStride;
  f2 += h * fStride;
  const int i = blockIdx.x * 256 + threadIdx.x;
  if (i >= rows) return;
  float s1 = 0.f, s2 = 0.f;
  for (int d = 0; d < D; ++d) {
    const float w = (float)Wh[(long)i * D + d];
    s1 += w * a[d];
    s2 += w * a[D + d];
  }
  f1[i] = s1;
  f2[i] = s2;
}

// ---------------------------------------------------------------------------
// Flash-style masked GAT attention layer.
//   out[i,:] = softmax_j( mask ? lrelu(f1_i+f2_j) : -9e15 ) @ Wh
// WhT is the PRE-TRANSPOSED value matrix [D][NV] (bf16).
// Block: 128 threads = 4 waves; 64 query rows; 64-key tiles, double-buffered.
// MODE 0: ELU -> bf16   MODE 1: ELU -> f32   MODE 2: raw f32
// ---------------------------------------------------------------------------
#define NEG_BIG (-9.0e15f)

template <int NF, int MODE>
__global__ __launch_bounds__(128) void gat_attention_kernel(
    const __bf16* __restrict__ WhT, const float* __restrict__ f1,
    const float* __restrict__ f2, const unsigned* __restrict__ maskbits,
    int NV,
    float* __restrict__ dstf, __bf16* __restrict__ dstb, int ldd, int colOff,
    long headStrideWh, int headStrideF, int headColStep) {
  constexpr int D = NF * 16;
  constexpr int WS = 72;            // LDS tile row stride (bf16): 144B
  __shared__ __align__(16) __bf16 Wt[2][64 * WS];   // [feature n][key k]
  __shared__ float f2s[2][64];
  __shared__ unsigned mw[2][128];   // 64 rows x 2 words

  const int head = blockIdx.y;
  WhT += (long)head * headStrideWh;
  f1  += head * headStrideF;
  f2  += head * headStrideF;
  const int co = colOff + head * headColStep;

  const int tid = threadIdx.x, lane = tid & 31, wave = tid >> 5;
  const int hi = lane >> 4, lr = lane & 15;
  const int i0 = blockIdx.x * 64;
  const int words = NV >> 5;
  const float f1v = f1[i0 + wave * 16 + lr];

  float m = -3.0e38f, l = 0.f;
  FragC acc[NF];
#pragma unroll
  for (int f = 0; f < NF; ++f)
#pragma unroll
    for (int r = 0; r < 8; ++r) acc[f].f[r] = 0.f;

  const uint4* WhTq = (const uint4*)WhT;
  const int NVq = NV >> 3;

  auto stage = [&](int j0, int b) {
    if (tid < 64) f2s[b][tid] = f2[j0 + tid];
    mw[b][tid] = maskbits[(long)(i0 + (tid >> 1)) * words + (j0 >> 5) + (tid & 1)];
    uint4* wtq = (uint4*)Wt[b];
#pragma unroll
    for (int idx = tid; idx < D * 8; idx += 128) {
      const int n = idx >> 3, c8 = idx & 7;
      copy16_g2l(WhTq + (long)n * NVq + (j0 >> 3) + c8, wtq + n * (WS / 8) + c8);
    }
  };

  stage(0, 0);
  async_wait_all();
  __syncthreads();

  int buf = 0;
  for (int j0 = 0; j0 < NV; j0 += 64) {
    if (j0 + 64 < NV) stage(j0 + 64, buf ^ 1);  // overlap with softmax + WMMA

    const unsigned w0 = mw[buf][(wave * 16 + lr) * 2 + 0];
    const unsigned w1 = mw[buf][(wave * 16 + lr) * 2 + 1];

    // e-values computed directly in A-fragment lane layout:
    // e<16 -> frag0 (j = hi*8 + {0..7,16..23}); e>=16 -> frag1 (+32)
    float ev[32];
    float tmax = -3.0e38f;
#pragma unroll
    for (int e = 0; e < 32; ++e) {
      const int joff = hi * 8 + (((e >> 1) & 3) << 1) + (e & 1) +
                       (((e >> 3) & 1) << 4) + ((e >> 4) << 5);
      const unsigned bit = (joff < 32) ? (w0 >> joff) : (w1 >> (joff - 32));
      const float s = f1v + f2s[buf][joff];
      const float el = s > 0.f ? s : 0.2f * s;     // leaky relu
      const float v = (bit & 1u) ? el : NEG_BIG;
      ev[e] = v;
      tmax = fmaxf(tmax, v);
    }
    // row state is split across lane pair (L, L+16): combine
    tmax = fmaxf(tmax, __shfl_xor(tmax, 16, 32));
    const float mnew = fmaxf(m, tmax);
    const float scale = __expf(m - mnew);

    float ts = 0.f;
    FragB pa0, pa1;
#pragma unroll
    for (int e = 0; e < 16; ++e) {
      const float p0 = __expf(ev[e] - mnew);
      const float p1 = __expf(ev[16 + e] - mnew);
      ts += p0 + p1;
      pa0.h[e] = (__bf16)p0;
      pa1.h[e] = (__bf16)p1;
    }
    ts += __shfl_xor(ts, 16, 32);
    l = l * scale + ts;
    m = mnew;

    // rescale accumulators: row of acc VGPR r is (hi*8 + r); scale in lane
    // (hi*8 + r)
    float sr[8];
#pragma unroll
    for (int r = 0; r < 8; ++r) sr[r] = __shfl(scale, hi * 8 + r, 32);
#pragma unroll
    for (int nf = 0; nf < NF; ++nf)
#pragma unroll
      for (int r = 0; r < 8; ++r) acc[nf].f[r] *= sr[r];

    // P @ Wh-tile : 64 keys -> two 16x16x32 WMMAs per 16-wide feature frag
#pragma unroll
    for (int nf = 0; nf < NF; ++nf) {
      FragB b0; b0.v = lds_bfrag(Wt[buf] + nf * 16 * WS, WS, lane);
      acc[nf].v = WMMA_BF16(pa0.v, b0.v, acc[nf].v);
      FragB b1; b1.v = lds_bfrag(Wt[buf] + nf * 16 * WS + 32, WS, lane);
      acc[nf].v = WMMA_BF16(pa1.v, b1.v, acc[nf].v);
    }
    async_wait_all();
    __syncthreads();
    buf ^= 1;
  }

  // normalize + epilogue
  const float linv = 1.0f / l;
  float lr8[8];
#pragma unroll
  for (int r = 0; r < 8; ++r) lr8[r] = __shfl(linv, hi * 8 + r, 32);

  const int col = lr;
#pragma unroll
  for (int nf = 0; nf < NF; ++nf) {
#pragma unroll
    for (int r = 0; r < 8; ++r) {
      const int row = i0 + wave * 16 + hi * 8 + r;
      float v = acc[nf].f[r] * lr8[r];
      if constexpr (MODE == 0) {
        v = v > 0.f ? v : __expf(v) - 1.f;  // ELU
        dstb[(long)row * ldd + co + nf * 16 + col] = (__bf16)v;
      } else if constexpr (MODE == 1) {
        v = v > 0.f ? v : __expf(v) - 1.f;  // ELU
        dstf[(long)row * ldd + co + nf * 16 + col] = v;
      } else {
        dstf[(long)row * ldd + nf * 16 + col] = v;
      }
    }
  }
}

// ---------------------------------------------------------------------------
// row-wise log_softmax over 16 classes
// ---------------------------------------------------------------------------
__global__ __launch_bounds__(256) void log_softmax16_kernel(const float* __restrict__ in,
                                                            float* __restrict__ out,
                                                            int rows) {
  const int i = blockIdx.x * 256 + threadIdx.x;
  if (i >= rows) return;
  float v[16];
  float mx = -3.0e38f;
#pragma unroll
  for (int c = 0; c < 16; ++c) { v[c] = in[i * 16 + c]; mx = fmaxf(mx, v[c]); }
  float s = 0.f;
#pragma unroll
  for (int c = 0; c < 16; ++c) s += __expf(v[c] - mx);
  const float ls = __logf(s);
#pragma unroll
  for (int c = 0; c < 16; ++c) out[i * 16 + c] = v[c] - mx - ls;
}

// ---------------------------------------------------------------------------
// z = out2 + original_x ; result2 = z@w_deg + b_deg ; result3 = z@W_deg3 + b_deg3
// ---------------------------------------------------------------------------
__global__ __launch_bounds__(128) void finalize_z_kernel(
    const float* __restrict__ out2r, const float* __restrict__ origx,
    const float* __restrict__ w_deg, const float* __restrict__ b_deg,
    const float* __restrict__ W_deg3, const float* __restrict__ b_deg3,
    float* __restrict__ res2, float* __restrict__ res3) {
  __shared__ float zs[64];
  const int i = blockIdx.x, tid = threadIdx.x;
  if (tid < 64) zs[tid] = out2r[(long)i * 64 + tid] + origx[(long)i * 64 + tid];
  __syncthreads();
  float s = b_deg3[tid];
#pragma unroll 8
  for (int k = 0; k < 64; ++k) s += zs[k] * W_deg3[k * 128 + tid];
  res3[(long)i * 128 + tid] = s;
  if (tid == 0) {
    float s2 = b_deg[0];
    for (int k = 0; k < 64; ++k) s2 += zs[k] * w_deg[k];
    res2[i] = s2;
  }
}

// ---------------------------------------------------------------------------
// Host launcher
// ---------------------------------------------------------------------------
extern "C" void kernel_launch(void* const* d_in, const int* in_sizes, int n_in,
                              void* d_out, int out_size, void* d_ws, size_t ws_size,
                              hipStream_t stream) {
  (void)in_sizes; (void)n_in; (void)out_size; (void)ws_size;
  constexpr int N = 4096, NFEAT = 512, NHID = 64, NCLASS = 16, NHEADS = 8;

  const float* x      = (const float*)d_in[0];
  const int*   adj    = (const int*)d_in[1];
  const float* W_att  = (const float*)d_in[2];
  const float* a_att  = (const float*)d_in[3];
  const float* W_out1 = (const float*)d_in[4];
  const float* a_out1 = (const float*)d_in[5];
  const float* W_out2 = (const float*)d_in[6];
  const float* a_out2 = (const float*)d_in[7];
  const float* W_down = (const float*)d_in[8];
  const float* b_down = (const float*)d_in[9];
  const float* w_deg  = (const float*)d_in[10];
  const float* b_deg  = (const float*)d_in[11];
  const float* W_deg3 = (const float*)d_in[12];
  const float* b_deg3 = (const float*)d_in[13];

  // ---- carve workspace (256B aligned) ----
  char* ws = (char*)d_ws;
  auto alloc = [&](size_t bytes) -> char* {
    char* p = ws;
    ws += (bytes + 255) & ~(size_t)255;
    return p;
  };
  __bf16*  xb     = (__bf16*)alloc((size_t)N * NFEAT * 2);
  __bf16*  WattT  = (__bf16*)alloc((size_t)NHEADS * NFEAT * NHID * 2);   // [h][64][512]
  __bf16*  WdownT = (__bf16*)alloc((size_t)NFEAT * NHID * 2);            // [64][512]
  __bf16*  Wout1T = (__bf16*)alloc((size_t)NFEAT * NCLASS * 2);          // [16][512]
  __bf16*  Wout2T = (__bf16*)alloc((size_t)NFEAT * NHID * 2);            // [64][512]
  __bf16*  WhH    = (__bf16*)alloc((size_t)NHEADS * N * NHID * 2);       // [h][4096][64]
  __bf16*  WhHT   = (__bf16*)alloc((size_t)NHEADS * N * NHID * 2);       // [h][64][4096]
  __bf16*  H      = (__bf16*)alloc((size_t)N * NFEAT * 2);               // [4096][512]
  __bf16*  Wh1b   = (__bf16*)alloc((size_t)N * NCLASS * 2);
  __bf16*  Wh1T   = (__bf16*)alloc((size_t)N * NCLASS * 2);              // [16][4096]
  __bf16*  Wh2b   = (__bf16*)alloc((size_t)N * NHID * 2);
  __bf16*  Wh2T   = (__bf16*)alloc((size_t)N * NHID * 2);                // [64][4096]
  float*   origx  = (float*)alloc((size_t)N * NHID * 4);
  float*   out1e  = (float*)alloc((size_t)N * NCLASS * 4);
  float*   out2r  = (float*)alloc((size_t)N * NHID * 4);
  float*   f1a    = (float*)alloc((size_t)10 * N * 4);
  float*   f2a    = (float*)alloc((size_t)10 * N * 4);
  unsigned* maskb = (unsigned*)alloc((size_t)N * (N / 32) * 4);

  float* res1 = (float*)d_out;
  float* res2 = res1 + (size_t)N * NCLASS;
  float* res3 = res2 + N;

  // 1) pack adjacency mask
  pack_mask_kernel<<<(N * N) / 256, 256, 0, stream>>>(adj, maskb);

  // 2) convert x; transpose+convert weights
  cvt_bf16_kernel<<<(N * NFEAT) / 256, 256, 0, stream>>>(x, xb, N * NFEAT);
  cvt_t_kernel<<<dim3((NFEAT * NHID) / 256, NHEADS), 256, 0, stream>>>(
      W_att, WattT, NFEAT, NHID, (long)NFEAT * NHID, (long)NFEAT * NHID);
  cvt_t_kernel<<<dim3((NFEAT * NHID) / 256, 1), 256, 0, stream>>>(
      W_down, WdownT, NFEAT, NHID, 0, 0);
  cvt_t_kernel<<<dim3((NFEAT * NCLASS) / 256, 1), 256, 0, stream>>>(
      W_out1, Wout1T, NFEAT, NCLASS, 0, 0);
  cvt_t_kernel<<<dim3((NFEAT * NHID) / 256, 1), 256, 0, stream>>>(
      W_out2, Wout2T, NFEAT, NHID, 0, 0);

  // 3) Wh[h] = x @ W_att[h] (batched over heads), bf16 out
  gemm_bf16_wmma<4><<<dim3(N / 128, NHEADS), 256, 0, stream>>>(
      xb, WattT, N, NFEAT, nullptr, WhH, nullptr,
      (long)NFEAT * NHID, (long)N * NHID);

  // 4) original_x = x @ W_down + b_down (f32 out)
  gemm_bf16_wmma<4><<<dim3(N / 128, 1), 256, 0, stream>>>(
      xb, WdownT, N, NFEAT, origx, nullptr, b_down, 0, 0);

  // 5) f1/f2 per head; transpose Wh per head
  compute_f12_kernel<<<dim3(N / 256, NHEADS), 256, 0, stream>>>(
      WhH, a_att, NHID, N, f1a, f2a, (long)N * NHID, 2 * NHID, N);
  transpose_bf16_kernel<NHID><<<dim3(N / 64, NHEADS), 256, 0, stream>>>(
      WhH, WhHT, N, (long)N * NHID, (long)N * NHID);

  // 6) head attention -> ELU -> H (bf16, col block per head)
  gat_attention_kernel<4, 0><<<dim3(N / 64, NHEADS), 128, 0, stream>>>(
      WhHT, f1a, f2a, maskb, N, nullptr, H, NFEAT, 0,
      (long)N * NHID, N, NHID);

  // 7) Wh1 = H @ W_out1 ; Wh2 = H @ W_out2 (bf16)
  gemm_bf16_wmma<1><<<dim3(N / 128, 1), 256, 0, stream>>>(
      H, Wout1T, N, NFEAT, nullptr, Wh1b, nullptr, 0, 0);
  gemm_bf16_wmma<4><<<dim3(N / 128, 1), 256, 0, stream>>>(
      H, Wout2T, N, NFEAT, nullptr, Wh2b, nullptr, 0, 0);

  // 8) f1/f2 + transposes for output layers
  compute_f12_kernel<<<dim3(N / 256, 1), 256, 0, stream>>>(
      Wh1b, a_out1, NCLASS, N, f1a + 8 * N, f2a + 8 * N, 0, 0, 0);
  compute_f12_kernel<<<dim3(N / 256, 1), 256, 0, stream>>>(
      Wh2b, a_out2, NHID, N, f1a + 9 * N, f2a + 9 * N, 0, 0, 0);
  transpose_bf16_kernel<NCLASS><<<dim3(N / 64, 1), 256, 0, stream>>>(
      Wh1b, Wh1T, N, 0, 0);
  transpose_bf16_kernel<NHID><<<dim3(N / 64, 1), 256, 0, stream>>>(
      Wh2b, Wh2T, N, 0, 0);

  // 9) output attention layers
  gat_attention_kernel<1, 1><<<dim3(N / 64, 1), 128, 0, stream>>>(
      Wh1T, f1a + 8 * N, f2a + 8 * N, maskb, N,
      out1e, nullptr, NCLASS, 0, 0, 0, 0);
  gat_attention_kernel<4, 2><<<dim3(N / 64, 1), 128, 0, stream>>>(
      Wh2T, f1a + 9 * N, f2a + 9 * N, maskb, N,
      out2r, nullptr, NHID, 0, 0, 0, 0);

  // 10) result1 = log_softmax(out1e)
  log_softmax16_kernel<<<N / 256, 256, 0, stream>>>(out1e, res1, N);

  // 11) z = out2 + original_x ; result2, result3
  finalize_z_kernel<<<N, 128, 0, stream>>>(out2r, origx, w_deg, b_deg,
                                           W_deg3, b_deg3, res2, res3);
}
